// TripletSemiHardLoss_81767587381278
// MI455X (gfx1250) — compile-verified, matrix operand order checked by reading, not run on hardware
//
#include <hip/hip_runtime.h>
#include <hip/hip_bf16.h>
#include <math.h>

typedef __attribute__((ext_vector_type(16))) __bf16 bf16x16;
typedef __attribute__((ext_vector_type(8)))  __bf16 bf16x8;
typedef __attribute__((ext_vector_type(4)))  __bf16 bf16x4;
typedef __attribute__((ext_vector_type(8)))  float  f32x8;

#define DIM       128
#define MARGIN    1.0f
#define PD_EPS    1e-6f
#define NORM_EPS  1e-12f

union BfTile { bf16x16 v; bf16x8 h[2]; };

// Load a 16x32 bf16 operand tile in the WMMA A layout (which also serves as the
// B operand layout for the symmetric product E*E^T, since B^T rows == E rows):
//   lane m (0-15):  K = k0+0..7   (VGPR0-3) and K = k0+16..23 (VGPR4-7)
//   lane m+16:      K = k0+8..15  and K = k0+24..31
__device__ inline bf16x16 load_ab_tile(const __bf16* __restrict__ base,
                                       int row0, int k0, int lane) {
  const int m  = lane & 15;
  const int kk = k0 + ((lane >> 4) << 3);
  const __bf16* p = base + (size_t)(row0 + m) * DIM + kk;
  BfTile t;
  t.h[0] = *(const bf16x8*)(p);
  t.h[1] = *(const bf16x8*)(p + 16);
  return t.v;
}

// Normalize rows; emit fp32 normalized E, bf16 hi/lo split, and per-row sq(|e|^2).
__global__ __launch_bounds__(256) void prep_kernel(
    const float* __restrict__ X, float* __restrict__ E,
    __bf16* __restrict__ Ehi, __bf16* __restrict__ Elo,
    float* __restrict__ sq, int n) {
  const int row = blockIdx.x * 8 + (threadIdx.x >> 5);
  if (row >= n) return;
  const int lane = threadIdx.x & 31;
  const float4 x = ((const float4*)(X + (size_t)row * DIM))[lane];
  float ss = x.x * x.x + x.y * x.y + x.z * x.z + x.w * x.w;
  #pragma unroll
  for (int s = 1; s < 32; s <<= 1) ss += __shfl_xor(ss, s, 32);
  const float inv = 1.0f / fmaxf(sqrtf(ss), NORM_EPS);
  float4 e = make_float4(x.x * inv, x.y * inv, x.z * inv, x.w * inv);
  ((float4*)(E + (size_t)row * DIM))[lane] = e;
  float ev[4] = {e.x, e.y, e.z, e.w};
  bf16x4 hi, lo;
  #pragma unroll
  for (int k = 0; k < 4; k++) {
    __bf16 h = (__bf16)ev[k];
    hi[k] = h;
    lo[k] = (__bf16)(ev[k] - (float)h);
  }
  *(bf16x4*)(Ehi + (size_t)row * DIM + lane * 4) = hi;
  *(bf16x4*)(Elo + (size_t)row * DIM + lane * 4) = lo;
  if (lane == 0) sq[row] = ss * inv * inv;
}

// Fused distance-tile GEMM (WMMA bf16, 3-term hi/lo split) + row-wise mining.
// POS_PASS=true : hardest positive (argmax over same-label, j!=i).
// POS_PASS=false: semi-hard negative (argmin over diff-label with d_ap<d<d_ap+M).
//
// Hot loop is shuffle-free: each lane privately tracks (best,idx) for its 8
// C-matrix rows over the single column it observes (jg = c0 + nn).  The 16-lane
// butterfly + cross-wave LDS combine run ONCE after the sweep.
template <bool POS_PASS>
__global__ __launch_bounds__(256) void mine_kernel(
    const __bf16* __restrict__ Ehi, const __bf16* __restrict__ Elo,
    const float* __restrict__ sq, const int* __restrict__ labels,
    const float* __restrict__ dap_in,
    float* __restrict__ best_out, int* __restrict__ idx_out, int n) {
  const int nt   = n >> 4;
  const int r0   = blockIdx.x << 4;
  const int lane = threadIdx.x & 31;
  const int wave = threadIdx.x >> 5;
  const int hh   = lane >> 4;     // lane half -> C row offset (+8)
  const int nn   = lane & 15;     // C column within tile

  __shared__ float s_sq[16];
  __shared__ int   s_lab[16];
  __shared__ float s_dap[16];
  __shared__ float s_best[8][16];
  __shared__ int   s_bidx[8][16];

  if (threadIdx.x < 16) {
    s_sq[threadIdx.x]  = sq[r0 + threadIdx.x];
    s_lab[threadIdx.x] = labels[r0 + threadIdx.x];
    s_dap[threadIdx.x] = POS_PASS ? 0.0f : dap_in[r0 + threadIdx.x];
  }
  __syncthreads();

  // A tile (16 rows x 128 K) resident in VGPRs for the whole column sweep.
  bf16x16 a_hi[4], a_lo[4];
  #pragma unroll
  for (int kc = 0; kc < 4; kc++) {
    a_hi[kc] = load_ab_tile(Ehi, r0, kc * 32, lane);
    a_lo[kc] = load_ab_tile(Elo, r0, kc * 32, lane);
  }

  float rsq[8]; int rlab[8]; float rdap[8];
  #pragma unroll
  for (int v = 0; v < 8; v++) {
    const int m = v + 8 * hh;
    rsq[v] = s_sq[m]; rlab[v] = s_lab[m]; rdap[v] = s_dap[m];
  }

  const float SENT = POS_PASS ? -__builtin_inff() : __builtin_inff();
  float best[8]; int bidx[8];
  #pragma unroll
  for (int v = 0; v < 8; v++) { best[v] = SENT; bidx[v] = 0; }

  for (int ct = wave; ct < nt; ct += 8) {
    const int c0 = ct << 4;
    const int ctn = ct + 8;
    if (ctn < nt) {  // hint next column tile toward the caches
      __builtin_prefetch(Ehi + (size_t)(ctn << 4) * DIM, 0, 0);
      __builtin_prefetch(Elo + (size_t)(ctn << 4) * DIM, 0, 0);
    }
    f32x8 c = {};
    #pragma unroll
    for (int kc = 0; kc < 4; kc++) {
      bf16x16 b_hi = load_ab_tile(Ehi, c0, kc * 32, lane);
      bf16x16 b_lo = load_ab_tile(Elo, c0, kc * 32, lane);
      c = __builtin_amdgcn_wmma_f32_16x16x32_bf16(false, a_hi[kc], false, b_hi,
                                                  (short)0, c, false, false);
      c = __builtin_amdgcn_wmma_f32_16x16x32_bf16(false, a_hi[kc], false, b_lo,
                                                  (short)0, c, false, false);
      c = __builtin_amdgcn_wmma_f32_16x16x32_bf16(false, a_lo[kc], false, b_hi,
                                                  (short)0, c, false, false);
    }
    const int   jg   = c0 + nn;
    const float csq  = sq[jg];
    const int   clab = labels[jg];
    #pragma unroll
    for (int v = 0; v < 8; v++) {
      const int ig = r0 + v + 8 * hh;
      const float d2   = rsq[v] + csq - 2.0f * c[v];
      const float dist = sqrtf(fmaxf(d2, 0.0f));
      bool sel;
      if (POS_PASS) sel = (clab == rlab[v]) && (jg != ig);
      else          sel = (clab != rlab[v]) && (dist > rdap[v]) &&
                          (dist < rdap[v] + MARGIN);
      const float score = sel ? dist : SENT;
      // jg is strictly increasing per lane, so strictly-better updates keep
      // the first (lowest-index) extremum, matching jnp.argmax/argmin.
      const bool take = POS_PASS ? (score > best[v]) : (score < best[v]);
      if (take) { best[v] = score; bidx[v] = jg; }
    }
  }

  // One-time 16-lane butterfly (stays within lane halves), min-index tie-break.
  #pragma unroll
  for (int v = 0; v < 8; v++) {
    float score = best[v]; int idx = bidx[v];
    #pragma unroll
    for (int s = 1; s < 16; s <<= 1) {
      const float od = __shfl_xor(score, s, 32);
      const int   oi = __shfl_xor(idx, s, 32);
      const bool take = POS_PASS ? (od > score || (od == score && oi < idx))
                                 : (od < score || (od == score && oi < idx));
      if (take) { score = od; idx = oi; }
    }
    best[v] = score; bidx[v] = idx;
  }

  if (nn == 0) {
    #pragma unroll
    for (int v = 0; v < 8; v++) {
      s_best[wave][v + 8 * hh] = best[v];
      s_bidx[wave][v + 8 * hh] = bidx[v];
    }
  }
  __syncthreads();
  if (threadIdx.x < 16) {
    const int m = threadIdx.x;
    float bb = s_best[0][m]; int bi = s_bidx[0][m];
    #pragma unroll
    for (int w = 1; w < 8; w++) {
      const float ob = s_best[w][m]; const int oi = s_bidx[w][m];
      const bool take = POS_PASS ? (ob > bb || (ob == bb && oi < bi))
                                 : (ob < bb || (ob == bb && oi < bi));
      if (take) { bb = ob; bi = oi; }
    }
    best_out[r0 + m] = bb;
    idx_out[r0 + m]  = bi;
  }
}

__global__ void init_acc(float* acc_sum, int* acc_cnt) {
  *acc_sum = 0.0f; *acc_cnt = 0;
}

// Gather (a,p,n) in fp32, TripletMarginLoss with pairwise_distance eps.
__global__ __launch_bounds__(256) void loss_kernel(
    const float* __restrict__ E,
    const int* __restrict__ pos_idx, const int* __restrict__ neg_idx,
    const float* __restrict__ posbest, const float* __restrict__ negbest,
    float* __restrict__ acc_sum, int* __restrict__ acc_cnt, int n) {
  const int row = blockIdx.x * 8 + (threadIdx.x >> 5);
  if (row >= n) return;
  const int lane = threadIdx.x & 31;
  const bool valid = (posbest[row] > -__builtin_inff()) &&
                     (negbest[row] <  __builtin_inff());
  const int pi = pos_idx[row], ni = neg_idx[row];
  const float4 a = ((const float4*)(E + (size_t)row * DIM))[lane];
  const float4 p = ((const float4*)(E + (size_t)pi  * DIM))[lane];
  const float4 q = ((const float4*)(E + (size_t)ni  * DIM))[lane];
  float s1 = 0.0f, s2 = 0.0f, d;
  d = a.x - p.x + PD_EPS; s1 += d * d;
  d = a.y - p.y + PD_EPS; s1 += d * d;
  d = a.z - p.z + PD_EPS; s1 += d * d;
  d = a.w - p.w + PD_EPS; s1 += d * d;
  d = a.x - q.x + PD_EPS; s2 += d * d;
  d = a.y - q.y + PD_EPS; s2 += d * d;
  d = a.z - q.z + PD_EPS; s2 += d * d;
  d = a.w - q.w + PD_EPS; s2 += d * d;
  #pragma unroll
  for (int s = 1; s < 32; s <<= 1) {
    s1 += __shfl_xor(s1, s, 32);
    s2 += __shfl_xor(s2, s, 32);
  }
  if (lane == 0 && valid) {
    const float hinge = fmaxf(sqrtf(s1) - sqrtf(s2) + MARGIN, 0.0f);
    atomicAdd(acc_sum, hinge);
    atomicAdd(acc_cnt, 1);
  }
}

__global__ void finalize(const float* acc_sum, const int* acc_cnt,
                         float* __restrict__ out) {
  const int c = *acc_cnt;
  out[0] = *acc_sum / (float)(c > 0 ? c : 1);  // 0/1 = 0 when no triplets
}

extern "C" void kernel_launch(void* const* d_in, const int* in_sizes, int n_in,
                              void* d_out, int out_size, void* d_ws, size_t ws_size,
                              hipStream_t stream) {
  const float* X      = (const float*)d_in[0];
  const int*   labels = (const int*)d_in[1];
  const int n = in_sizes[1];              // 8192 embeddings, DIM=128

  char* ws = (char*)d_ws;
  size_t off = 0;
  auto carve = [&](size_t bytes) -> void* {
    off = (off + 255) & ~(size_t)255;
    void* p = ws + off; off += bytes; return p;
  };
  float*  E    = (float*)carve((size_t)n * DIM * sizeof(float));
  __bf16* Ehi  = (__bf16*)carve((size_t)n * DIM * sizeof(__bf16));
  __bf16* Elo  = (__bf16*)carve((size_t)n * DIM * sizeof(__bf16));
  float*  sq   = (float*)carve((size_t)n * sizeof(float));
  float*  posb = (float*)carve((size_t)n * sizeof(float));
  int*    posi = (int*)carve((size_t)n * sizeof(int));
  float*  negb = (float*)carve((size_t)n * sizeof(float));
  int*    negi = (int*)carve((size_t)n * sizeof(int));
  float*  accs = (float*)carve(sizeof(float));
  int*    accc = (int*)carve(sizeof(int));
  (void)ws_size; (void)n_in; (void)out_size;

  prep_kernel<<<(n + 7) / 8, 256, 0, stream>>>(X, E, Ehi, Elo, sq, n);
  mine_kernel<true ><<<n / 16, 256, 0, stream>>>(Ehi, Elo, sq, labels, nullptr,
                                                 posb, posi, n);
  mine_kernel<false><<<n / 16, 256, 0, stream>>>(Ehi, Elo, sq, labels, posb,
                                                 negb, negi, n);
  init_acc<<<1, 1, 0, stream>>>(accs, accc);
  loss_kernel<<<(n + 7) / 8, 256, 0, stream>>>(E, posi, negi, posb, negb,
                                               accs, accc, n);
  finalize<<<1, 1, 0, stream>>>(accs, accc, (float*)d_out);
}
